// AttentionalPositionEncoding_65601330479499
// MI455X (gfx1250) — matrix operand order repaired and checked by reading, not compiled, not run on hardware
//
#include <hip/hip_runtime.h>
#include <hip/hip_bf16.h>

// ---------------------------------------------------------------------------
// AttentionalPositionEncoding for MI455X (gfx1250), wave32 + WMMA f16->f32.
// All matmuls use v_wmma_f32_16x16x32_f16.  ~15 GFLOP over ~25MB of
// L2-resident data => compute-bound; tiles sized for operand reuse:
//   GEMM: 32x64 per wave (8 wmma / 12 b128 loads per K-step)
//   Attn: 32 queries x 32 keys per wave iter (8 wmma / 8 b128 loads)
// ---------------------------------------------------------------------------

typedef __attribute__((ext_vector_type(16))) _Float16 v16h;
typedef __attribute__((ext_vector_type(8)))  _Float16 v8h;
typedef __attribute__((ext_vector_type(8)))  float    v8f;

#define DMODEL 256
#define NTOK   1024     // 32*32
#define BATCH  8
#define HEADS  8
#define DK     32
#define MTOT   (BATCH * NTOK)   // 8192

// -------------------- lane-layout load helpers -----------------------------
// A-matrix (16x32 f16, row-major src, leading dim ld):
//   lanes 0-15 : row M=lane,     K in {0..15}
//   lanes 16-31: row M=lane-16,  K in {16..31}
static __device__ inline v16h load_a_global(const _Float16* __restrict__ base,
                                            int ld, int row0, int k0,
                                            int r, int hi) {
  const _Float16* p0 = base + (size_t)(row0 + r) * ld + k0 + hi * 16;
  v8h c0 = *(const v8h*)(p0);
  v8h c1 = *(const v8h*)(p0 + 8);
  v16h a;
#pragma unroll
  for (int i = 0; i < 8; ++i) { a[i] = c0[i]; a[i + 8] = c1[i]; }
  return a;
}

static __device__ inline v16h load_a_lds(const _Float16* p0) {
  v8h c0 = *(const v8h*)(p0);
  v8h c1 = *(const v8h*)(p0 + 8);
  v16h a;
#pragma unroll
  for (int i = 0; i < 8; ++i) { a[i] = c0[i]; a[i + 8] = c1[i]; }
  return a;
}

// B-matrix (32x16 f16): lane holds one COLUMN of B = 16 contiguous f16 when
// the column is contiguous in memory.  ptr must point at that column start.
static __device__ inline v16h load_b_contig(const _Float16* __restrict__ p) {
  return *(const v16h*)p;
}

// -------------------- prep kernels -----------------------------------------
__global__ void cvt_w_f16(const float* __restrict__ s, _Float16* __restrict__ d) {
  int i = blockIdx.x * blockDim.x + threadIdx.x;
  if (i < DMODEL * DMODEL) d[i] = (_Float16)s[i];
}

// src [B][256][1024] (f32, channel-major) -> dst [B*1024][256] (f16, token-major)
__global__ void transpose_cvt(const float* __restrict__ src, _Float16* __restrict__ dst) {
  __shared__ float t[16][17];
  int b  = blockIdx.z;
  int n0 = blockIdx.x * 16;
  int c0 = blockIdx.y * 16;
  int tx = threadIdx.x, ty = threadIdx.y;
  t[ty][tx] = src[((size_t)b * DMODEL + c0 + ty) * NTOK + n0 + tx];
  __syncthreads();
  dst[((size_t)b * NTOK + n0 + ty) * DMODEL + c0 + tx] = (_Float16)t[tx][ty];
}

// -------------------- generic projection GEMM -------------------------------
// out = A[M,256] @ W^T (+bias) (+epilogue).  One wave computes a 32x64 tile.
// mode 0: store f16 row-major [M,256]
// mode 1: store f16 transposed-per-feature [B][256][1024]  (for V)
// mode 2: += aux (f16 [M,256]), store f16 row-major         (for Wf + attn)
// mode 3: += residual from x [B][256][1024], store f32 out  (final Wo)
__global__ __launch_bounds__(32)
void gemm_wmma(const _Float16* __restrict__ A, const _Float16* __restrict__ W,
               const float* __restrict__ bias, const _Float16* __restrict__ aux,
               const float* __restrict__ xres, void* __restrict__ outp,
               int M, int mode) {
  int lane = threadIdx.x & 31;
  int r  = lane & 15;
  int hi = lane >> 4;

  int tile = blockIdx.x;
  int dt = tile & 3;          // 4 tiles of 64 cols across DMODEL=256
  int mt = tile >> 2;
  int m0 = mt * 32;
  int d0 = dt * 64;

  v8f c[2][4];
#pragma unroll
  for (int i = 0; i < 2; ++i)
#pragma unroll
    for (int j = 0; j < 4; ++j) c[i][j] = (v8f){};

#pragma unroll
  for (int kk = 0; kk < 8; ++kk) {
    int k0 = kk * 32;
    if (kk < 7) {
      __builtin_prefetch(A + (size_t)(m0 + r) * DMODEL + k0 + 32, 0, 1);
      __builtin_prefetch(W + (size_t)(d0 + r) * DMODEL + k0 + 32, 0, 1);
    }
    v16h a0 = load_a_global(A, DMODEL, m0,      k0, r, hi);
    v16h a1 = load_a_global(A, DMODEL, m0 + 16, k0, r, hi);
#pragma unroll
    for (int j = 0; j < 4; ++j) {
      // B = W^T tile: column (d0+16j+r) of W^T == row (d0+16j+r) of W.
      const _Float16* bp = W + (size_t)(d0 + 16 * j + r) * DMODEL + k0 + hi * 16;
      v16h bmat = load_b_contig(bp);
      c[0][j] = __builtin_amdgcn_wmma_f32_16x16x32_f16(
          false, a0, false, bmat, (short)0, c[0][j], false, false);
      c[1][j] = __builtin_amdgcn_wmma_f32_16x16x32_f16(
          false, a1, false, bmat, (short)0, c[1][j], false, false);
    }
  }

#pragma unroll
  for (int i = 0; i < 2; ++i) {
    int mbase = m0 + 16 * i;
    int b  = mbase >> 10;     // batch index (tiles never cross batches)
    int n0 = mbase & 1023;

#pragma unroll
    for (int j = 0; j < 4; ++j) {
      int dglob = d0 + 16 * j + r;
      float bv = bias ? bias[dglob] : 0.0f;

      if (mode == 1) {
        // transposed store: v_t[b][dglob][n], contiguous 8 f16 per lane
        v8h pk;
#pragma unroll
        for (int vi = 0; vi < 8; ++vi) pk[vi] = (_Float16)(c[i][j][vi] + bv);
        _Float16* o = (_Float16*)outp;
        *(v8h*)(o + ((size_t)b * DMODEL + dglob) * NTOK + n0 + hi * 8) = pk;
      } else if (mode == 3) {
        float* o = (float*)outp;
        const float* rp = xres + ((size_t)b * DMODEL + dglob) * NTOK + n0 + hi * 8;
#pragma unroll
        for (int vi = 0; vi < 8; ++vi) {
          int m = mbase + vi + hi * 8;
          o[(size_t)m * DMODEL + dglob] = c[i][j][vi] + bv + rp[vi];
        }
      } else {
        _Float16* o = (_Float16*)outp;
#pragma unroll
        for (int vi = 0; vi < 8; ++vi) {
          int m = mbase + vi + hi * 8;
          float val = c[i][j][vi] + bv;
          if (mode == 2) val += (float)aux[(size_t)m * DMODEL + dglob];
          o[(size_t)m * DMODEL + dglob] = (_Float16)val;
        }
      }
    }
  }
}

// -------------------- flash attention: (b,h, 32-query tile) per wave --------
// Two 16-row query tiles share every K/V operand; online softmax does one
// row per lane (32 rows == 32 lanes, no divergence anywhere).
__global__ __launch_bounds__(32)
void attn_wmma(const _Float16* __restrict__ q, const _Float16* __restrict__ k,
               const _Float16* __restrict__ vt, _Float16* __restrict__ attn) {
  __shared__ float    sc[32 * 32];
  __shared__ _Float16 pb[32 * 32];
  __shared__ float    mrow[32], lrow[32], alph[32];

  int lane = threadIdx.x & 31;
  int r  = lane & 15;
  int hi = lane >> 4;

  int qt = blockIdx.x & 31;          // 32 query-pair tiles of 32 rows
  int h  = (blockIdx.x >> 5) & 7;
  int b  = blockIdx.x >> 8;
  int q0 = qt * 32;

  const float SCALE = 0.17677669529663687f;   // 1/sqrt(32)

  mrow[lane] = -1e30f;
  lrow[lane] = 0.0f;
  __syncthreads();

  // Q tiles: 2 x (16 queries x dk=32) A operands, loaded once.
  const _Float16* qbase = q + (size_t)(b * NTOK) * DMODEL + h * DK;
  v16h qa0 = load_a_lds(qbase + (size_t)(q0 + r)      * DMODEL + hi * 16);
  v16h qa1 = load_a_lds(qbase + (size_t)(q0 + 16 + r) * DMODEL + hi * 16);

  const _Float16* kb  = k  + (size_t)(b * NTOK) * DMODEL + h * DK;
  const _Float16* vtb = vt + ((size_t)b * DMODEL + h * DK) * NTOK;

  v8f o00 = (v8f){}, o01 = (v8f){}, o10 = (v8f){}, o11 = (v8f){};

  for (int it = 0; it < 32; ++it) {
    int key0 = it * 32;
    // scores: B = K^T, lane column = key, contiguous dk chunk of K row
    v16h bk0 = load_b_contig(kb + (size_t)(key0 + r) * DMODEL + hi * 16);
    v16h bk1 = load_b_contig(kb + (size_t)(key0 + 16 + r) * DMODEL + hi * 16);
    v8f z = (v8f){};
    v8f s00 = __builtin_amdgcn_wmma_f32_16x16x32_f16(false, qa0, false, bk0,
                                                     (short)0, z, false, false);
    v8f s01 = __builtin_amdgcn_wmma_f32_16x16x32_f16(false, qa0, false, bk1,
                                                     (short)0, z, false, false);
    v8f s10 = __builtin_amdgcn_wmma_f32_16x16x32_f16(false, qa1, false, bk0,
                                                     (short)0, z, false, false);
    v8f s11 = __builtin_amdgcn_wmma_f32_16x16x32_f16(false, qa1, false, bk1,
                                                     (short)0, z, false, false);
#pragma unroll
    for (int vi = 0; vi < 8; ++vi) {
      int row0 = vi + hi * 8;
      sc[row0 * 32 + r]             = s00[vi] * SCALE;
      sc[row0 * 32 + r + 16]        = s01[vi] * SCALE;
      sc[(row0 + 16) * 32 + r]      = s10[vi] * SCALE;
      sc[(row0 + 16) * 32 + r + 16] = s11[vi] * SCALE;
    }
    __syncthreads();

    // online softmax: one row per lane, all 32 lanes active
    {
      float mt = -1e30f;
#pragma unroll
      for (int j = 0; j < 32; ++j) mt = fmaxf(mt, sc[lane * 32 + j]);
      float mnew = fmaxf(mrow[lane], mt);
      float al = __expf(mrow[lane] - mnew);
      float sum = 0.0f;
#pragma unroll
      for (int j = 0; j < 32; ++j) {
        float pj = __expf(sc[lane * 32 + j] - mnew);
        pb[lane * 32 + j] = (_Float16)pj;
        sum += pj;
      }
      lrow[lane] = lrow[lane] * al + sum;
      mrow[lane] = mnew;
      alph[lane] = al;
    }
    __syncthreads();

    // rescale running output by alpha(row)
#pragma unroll
    for (int vi = 0; vi < 8; ++vi) {
      float a0 = alph[vi + hi * 8];
      float a1 = alph[16 + vi + hi * 8];
      o00[vi] *= a0; o01[vi] *= a0;
      o10[vi] *= a1; o11[vi] *= a1;
    }
    // PV: A = P (from LDS), B = V chunk via transposed V (contiguous)
    v16h pa0 = load_a_lds(pb + r * 32 + hi * 16);
    v16h pa1 = load_a_lds(pb + (16 + r) * 32 + hi * 16);
    v16h bv0 = load_b_contig(vtb + (size_t)(r)      * NTOK + key0 + hi * 16);
    v16h bv1 = load_b_contig(vtb + (size_t)(r + 16) * NTOK + key0 + hi * 16);
    o00 = __builtin_amdgcn_wmma_f32_16x16x32_f16(false, pa0, false, bv0,
                                                 (short)0, o00, false, false);
    o01 = __builtin_amdgcn_wmma_f32_16x16x32_f16(false, pa0, false, bv1,
                                                 (short)0, o01, false, false);
    o10 = __builtin_amdgcn_wmma_f32_16x16x32_f16(false, pa1, false, bv0,
                                                 (short)0, o10, false, false);
    o11 = __builtin_amdgcn_wmma_f32_16x16x32_f16(false, pa1, false, bv1,
                                                 (short)0, o11, false, false);
  }

  // normalize and store attn_out f16 row-major [B*N, 256]
#pragma unroll
  for (int vi = 0; vi < 8; ++vi) {
    int row0 = vi + hi * 8;
    float li0 = 1.0f / lrow[row0];
    float li1 = 1.0f / lrow[row0 + 16];
    int m0g = b * NTOK + q0 + row0;
    attn[(size_t)m0g * DMODEL + h * DK + r]      = (_Float16)(o00[vi] * li0);
    attn[(size_t)m0g * DMODEL + h * DK + r + 16] = (_Float16)(o01[vi] * li0);
    attn[(size_t)(m0g + 16) * DMODEL + h * DK + r]      = (_Float16)(o10[vi] * li1);
    attn[(size_t)(m0g + 16) * DMODEL + h * DK + r + 16] = (_Float16)(o11[vi] * li1);
  }
}

// -------------------- launch ------------------------------------------------
extern "C" void kernel_launch(void* const* d_in, const int* in_sizes, int n_in,
                              void* d_out, int out_size, void* d_ws, size_t ws_size,
                              hipStream_t stream) {
  (void)in_sizes; (void)n_in; (void)out_size; (void)ws_size;

  const float* x   = (const float*)d_in[0];
  const float* pos = (const float*)d_in[1];
  const float* Wq  = (const float*)d_in[2];
  const float* bq  = (const float*)d_in[3];
  const float* Wk  = (const float*)d_in[4];
  const float* bk  = (const float*)d_in[5];
  const float* Wv  = (const float*)d_in[6];
  const float* bv  = (const float*)d_in[7];
  const float* Wo  = (const float*)d_in[8];
  const float* bo  = (const float*)d_in[9];
  const float* Wc  = (const float*)d_in[10];
  const float* Wf  = (const float*)d_in[11];
  const float* bf  = (const float*)d_in[12];

  char* ws = (char*)d_ws;
  const size_t WSZ = (size_t)DMODEL * DMODEL * sizeof(_Float16);   // 128 KB
  const size_t TSZ = (size_t)MTOT * DMODEL * sizeof(_Float16);     // 4 MB
  _Float16* wc16 = (_Float16*)(ws + 0 * WSZ);
  _Float16* wq16 = (_Float16*)(ws + 1 * WSZ);
  _Float16* wk16 = (_Float16*)(ws + 2 * WSZ);
  _Float16* wv16 = (_Float16*)(ws + 3 * WSZ);
  _Float16* wf16 = (_Float16*)(ws + 4 * WSZ);
  _Float16* wo16 = (_Float16*)(ws + 5 * WSZ);
  char* bufs = ws + 6 * WSZ;
  _Float16* xt16  = (_Float16*)(bufs + 0 * TSZ);   // also reused as u16
  _Float16* pt16  = (_Float16*)(bufs + 1 * TSZ);   // also reused as attn16
  _Float16* cnn16 = (_Float16*)(bufs + 2 * TSZ);
  _Float16* q16   = (_Float16*)(bufs + 3 * TSZ);
  _Float16* k16   = (_Float16*)(bufs + 4 * TSZ);
  _Float16* vt16  = (_Float16*)(bufs + 5 * TSZ);
  _Float16* u16   = xt16;   // x tokens dead after cnn GEMM
  _Float16* at16  = pt16;   // pos tokens dead after k/v GEMMs

  // 1. weight conversion
  cvt_w_f16<<<256, 256, 0, stream>>>(Wc, wc16);
  cvt_w_f16<<<256, 256, 0, stream>>>(Wq, wq16);
  cvt_w_f16<<<256, 256, 0, stream>>>(Wk, wk16);
  cvt_w_f16<<<256, 256, 0, stream>>>(Wv, wv16);
  cvt_w_f16<<<256, 256, 0, stream>>>(Wf, wf16);
  cvt_w_f16<<<256, 256, 0, stream>>>(Wo, wo16);

  // 2. token-major f16 copies of x and pos
  dim3 tb(16, 16), tg(NTOK / 16, DMODEL / 16, BATCH);
  transpose_cvt<<<tg, tb, 0, stream>>>(x,   xt16);
  transpose_cvt<<<tg, tb, 0, stream>>>(pos, pt16);

  const int GB = (MTOT / 32) * 4;   // 1024 waves, one 32x64 tile each

  // 3. cnn = x_tok @ Wc^T
  gemm_wmma<<<GB, 32, 0, stream>>>(xt16, wc16, nullptr, nullptr, nullptr,
                                   cnn16, MTOT, 0);
  // 4. q = cnn @ Wq^T + bq
  gemm_wmma<<<GB, 32, 0, stream>>>(cnn16, wq16, bq, nullptr, nullptr,
                                   q16, MTOT, 0);
  // 5. k = pos @ Wk^T + bk
  gemm_wmma<<<GB, 32, 0, stream>>>(pt16, wk16, bk, nullptr, nullptr,
                                   k16, MTOT, 0);
  // 6. v = pos @ Wv^T + bv, stored transposed per feature
  gemm_wmma<<<GB, 32, 0, stream>>>(pt16, wv16, bv, nullptr, nullptr,
                                   vt16, MTOT, 1);
  // 7. attention: 2048 waves, 32 queries each
  attn_wmma<<<BATCH * HEADS * (NTOK / 32), 32, 0, stream>>>(q16, k16, vt16, at16);
  // 8. u = cnn @ Wf^T + bf + attn_out
  gemm_wmma<<<GB, 32, 0, stream>>>(cnn16, wf16, bf, at16, nullptr,
                                   u16, MTOT, 2);
  // 9. out = u @ Wo^T + bo + x_tok   (f32)
  gemm_wmma<<<GB, 32, 0, stream>>>(u16, wo16, bo, nullptr, x,
                                   d_out, MTOT, 3);
}
